// DynamicMemory_9474697855695
// MI455X (gfx1250) — matrix operand order repaired and checked by reading, not compiled, not run on hardware
//
#include <hip/hip_runtime.h>

// Problem constants (reference: T,B,H,M = 128,512,256,64)
#define T_STEPS 128
#define B_SZ    512
#define H_SZ    256
#define M_SZ    64
#define EPSV    1e-12f

typedef __bf16        v16bf  __attribute__((ext_vector_type(16)));
typedef float         v8f    __attribute__((ext_vector_type(8)));
typedef unsigned int  u32x4  __attribute__((ext_vector_type(4)));
typedef int           i32x8t __attribute__((ext_vector_type(8)));
typedef int           i32x4t __attribute__((ext_vector_type(4)));

#if defined(__gfx1250__) && __has_builtin(__builtin_amdgcn_tensor_load_to_lds) && \
    __has_builtin(__builtin_amdgcn_s_wait_tensorcnt)
#define USE_TDM 1
#else
#define USE_TDM 0
#endif

union Frag {
  v16bf          bf;
  u32x4          q[2];
  unsigned short us[16];
};

__device__ __forceinline__ unsigned short f2bf(float f) {
  unsigned int u = __float_as_uint(f);
  u += 0x7FFFu + ((u >> 16) & 1u);   // round-to-nearest-even
  return (unsigned short)(u >> 16);
}
__device__ __forceinline__ float bf2f(unsigned short s) {
  return __uint_as_float(((unsigned int)s) << 16);
}
// Packed f32x2 -> bf16x2 (one v_cvt_pk_bf16_f32 when available)
__device__ __forceinline__ unsigned cvt2bf(float lo, float hi) {
#if defined(__gfx1250__) && __has_builtin(__builtin_amdgcn_cvt_pk_bf16_f32)
  auto r = __builtin_amdgcn_cvt_pk_bf16_f32(lo, hi);
  return __builtin_bit_cast(unsigned, r);
#else
  return (unsigned)f2bf(lo) | ((unsigned)f2bf(hi) << 16);
#endif
}

// ---------------------------------------------------------------------------
// Kernel 1: KC[m,n] = sum_k keys[m,k] * V[k,n]   (tiny: 64x256x256, VALU)
// ---------------------------------------------------------------------------
__global__ __launch_bounds__(256) void kc_kernel(const float* __restrict__ keys,
                                                 const float* __restrict__ V,
                                                 float* __restrict__ KC) {
  const int m = blockIdx.x;
  const int n = threadIdx.x;
  float acc = 0.f;
  for (int k = 0; k < H_SZ; ++k)
    acc = fmaf(keys[m * H_SZ + k], V[k * H_SZ + n], acc);
  KC[m * H_SZ + n] = acc;
}

// ---------------------------------------------------------------------------
// Kernel 2: SW[t*B+b, n] = sum_k stories[t*B+b, k] * W[k, n]
// One big (T*B=65536) x 256 x 256 bf16 WMMA GEMM. 512 WGs x 128 rows.
// LDS: W transposed to bf16 ( wt[n*256+k] ), 128 KB.
// ---------------------------------------------------------------------------
__global__ __launch_bounds__(256) void sw_gemm_kernel(const float* __restrict__ stories,
                                                      const float* __restrict__ W,
                                                      float* __restrict__ SW) {
  extern __shared__ char smem[];
  unsigned short* wt = (unsigned short*)smem;   // [256][256] bf16, wt[n][k] = W[k][n]
  const int tid = threadIdx.x;

  // Stage W^T as bf16: thread tid owns source row k = tid.
  {
    const float4* wrow = (const float4*)(W + tid * H_SZ);
    for (int n4 = 0; n4 < 64; ++n4) {
      float4 v = wrow[n4];
      const int n = n4 * 4;
      wt[(n + 0) * H_SZ + tid] = f2bf(v.x);
      wt[(n + 1) * H_SZ + tid] = f2bf(v.y);
      wt[(n + 2) * H_SZ + tid] = f2bf(v.z);
      wt[(n + 3) * H_SZ + tid] = f2bf(v.w);
    }
  }
  __syncthreads();

  const int wave   = tid >> 5;
  const int lane   = tid & 31;
  const int laneLo = lane & 15;
  const int laneHi = (lane >> 4) & 1;
  const int rowA   = blockIdx.x * 128 + wave * 16 + laneLo;

  const v8f zero8 = {0.f, 0.f, 0.f, 0.f, 0.f, 0.f, 0.f, 0.f};
  v8f acc[16];
#pragma unroll
  for (int i = 0; i < 16; ++i) acc[i] = zero8;

  for (int kt = 0; kt < 8; ++kt) {
    // A fragment (16x32 bf16): lanes 0-15 rows, K = kbase..+15
    Frag a;
    const int    kbase = kt * 32 + laneHi * 16;
    const float* ap    = stories + rowA * H_SZ + kbase;
    const float4 a0 = *(const float4*)(ap);
    const float4 a1 = *(const float4*)(ap + 4);
    const float4 a2 = *(const float4*)(ap + 8);
    const float4 a3 = *(const float4*)(ap + 12);
    a.q[0] = (u32x4){cvt2bf(a0.x, a0.y), cvt2bf(a0.z, a0.w),
                     cvt2bf(a1.x, a1.y), cvt2bf(a1.z, a1.w)};
    a.q[1] = (u32x4){cvt2bf(a2.x, a2.y), cvt2bf(a2.z, a2.w),
                     cvt2bf(a3.x, a3.y), cvt2bf(a3.z, a3.w)};

    const int kb16 = kt * 32 + laneHi * 16;
#pragma unroll
    for (int nt = 0; nt < 16; ++nt) {
      Frag b;
      const unsigned short* bp = wt + (nt * 16 + laneLo) * H_SZ + kb16;
      b.q[0] = *(const u32x4*)(bp);
      b.q[1] = *(const u32x4*)(bp + 8);
      acc[nt] = __builtin_amdgcn_wmma_f32_16x16x32_bf16(false, a.bf, false, b.bf,
                                                        (short)0, acc[nt], false, false);
    }
  }

  const int rowC0 = blockIdx.x * 128 + wave * 16 + laneHi * 8;
#pragma unroll
  for (int nt = 0; nt < 16; ++nt) {
    const int col = nt * 16 + laneLo;
#pragma unroll
    for (int j = 0; j < 8; ++j)
      SW[(rowC0 + j) * H_SZ + col] = acc[nt][j];
  }
}

// ---------------------------------------------------------------------------
// Kernel 3: recurrent scan. 256 WGs, each owns all M=64 slots for 2 batch
// columns -> 128 state rows of H=256. State tile in LDS as bf16 (it IS the
// WMMA A operand), U^T in LDS bf16. Per step: async global->LDS staging,
// gate dots (VALU), 128x256x256 bf16 WMMA GEMM per WG, gated PReLU update +
// row L2-normalize in C-register layout. keys staged once via TDM.
// ---------------------------------------------------------------------------
__global__ __launch_bounds__(256) void scan_kernel(const float* __restrict__ stories,
                                                   const float* __restrict__ mask,
                                                   const float* __restrict__ keys,
                                                   const float* __restrict__ U,
                                                   const float* __restrict__ prelu_a,
                                                   const float* __restrict__ SW,
                                                   const float* __restrict__ KC,
                                                   float* __restrict__ memOut,
                                                   float* __restrict__ gatesOut) {
  extern __shared__ char smem[];
  unsigned short* ut    = (unsigned short*)smem;             // 131072 B : U^T bf16 [n][k]
  unsigned short* memB  = (unsigned short*)(smem + 131072);  //  65536 B : state bf16 [r=2m+bl][h]
  float*          sentF = (float*)(smem + 196608);           //   2048 B : sent [2][256]
  float*          swF   = (float*)(smem + 198656);           //   2048 B : sent@W [2][256]
  float*          partF = (float*)(smem + 200704);           //   1024 B : dot partials
  float*          gateF = (float*)(smem + 201728);           //    512 B : gate*mask per row
  float*          ssF   = (float*)(smem + 202240);           //    512 B : row sum-of-squares

  const int tid = threadIdx.x;
  const int bg0 = blockIdx.x * 2;   // this WG's two batch columns

  // ---- one-time: pull keys (64 KB, contiguous) into LDS scratch via the
  //      Tensor Data Mover, then convert to the bf16 state tile ----
#if USE_TDM
  if ((tid >> 5) == 0) {   // one wave issues the DMA (TENSORcnt-tracked)
    const unsigned long long ga = (unsigned long long)(uintptr_t)keys;
    const unsigned ldsBase = (unsigned)(uintptr_t)smem;   // scratch = ut region
    // D# group0: count=1, lds_addr, global_addr[56:0], type=2
    u32x4 g0 = {1u, ldsBase, (unsigned)ga,
                (unsigned)(((ga >> 32) & 0x1FFFFFFull) | (2u << 30))};
    // D# group1: data_size=4B(code 2); tensor_dim0=tile_dim0=16384; dim1=1;
    //            tensor_dim0_stride=16384
    i32x8t g1 = {(int)(2u << 16), (int)0x40000000, 0x00010000, (int)0x40000000,
                 1, 16384, 0, 0};
    i32x4t gz4 = {0, 0, 0, 0};
    i32x8t gz8 = {0, 0, 0, 0, 0, 0, 0, 0};
    __builtin_amdgcn_tensor_load_to_lds(g0, g1, gz4, gz4, gz8, 0);
    __builtin_amdgcn_s_wait_tensorcnt(0);
  }
  __syncthreads();
  const float* ksrc = (const float*)smem;   // keys now in LDS
#else
  const float* ksrc = keys;
#endif
  // init state rows = keys[m] broadcast (packed bf16 pairs)
  {
    unsigned* md = (unsigned*)memB;
    for (int idx = tid; idx < 128 * 128; idx += 256) {
      const int r = idx >> 7, hp = idx & 127;
      const int m = r >> 1;
      md[r * 128 + hp] = cvt2bf(ksrc[m * H_SZ + hp * 2], ksrc[m * H_SZ + hp * 2 + 1]);
    }
  }
  __syncthreads();   // keys scratch consumed; ut region free to overwrite

  // Stage U^T bf16 (thread tid owns source row k = tid; transposed dest -> scalar writes)
  {
    const float4* urow = (const float4*)(U + tid * H_SZ);
    for (int n4 = 0; n4 < 64; ++n4) {
      float4 v = urow[n4];
      const int n = n4 * 4;
      ut[(n + 0) * H_SZ + tid] = f2bf(v.x);
      ut[(n + 1) * H_SZ + tid] = f2bf(v.y);
      ut[(n + 2) * H_SZ + tid] = f2bf(v.z);
      ut[(n + 3) * H_SZ + tid] = f2bf(v.w);
    }
  }
  __syncthreads();

  const int wave = tid >> 5, lane = tid & 31;
  const int laneLo = lane & 15, laneHi = (lane >> 4) & 1;
  const int rDot = tid >> 1, halfDot = tid & 1;
  const int mDot = rDot >> 1, blDot = rDot & 1;
  const v8f zero8 = {0.f, 0.f, 0.f, 0.f, 0.f, 0.f, 0.f, 0.f};

  // Per-thread constants for the per-step async staging (one b128 per thread):
  // threads 0..127 stage sent, 128..255 stage sent@W.
  const int sSel = tid >> 7;
  const int sC4  = tid & 127;
  const int sBl  = sC4 >> 6;
  const int sH   = (sC4 & 63) * 4;
  const float* sGlobalBase = (sSel ? SW : stories) + (bg0 + sBl) * H_SZ + sH;
  float*       sLdsDst     = (sSel ? swF : sentF) + sC4 * 4;
  const unsigned sLdsOff   = (unsigned)(uintptr_t)sLdsDst;

  for (int t = 0; t < T_STEPS; ++t) {
    // ---- async global -> LDS staging of sent / sent@W (ASYNCcnt path) ----
    {
      const float* gsrc = sGlobalBase + t * (B_SZ * H_SZ);
      const unsigned long long ga = (unsigned long long)(uintptr_t)gsrc;
      asm volatile("global_load_async_to_lds_b128 %0, %1, off"
                   :: "v"(sLdsOff), "v"(ga) : "memory");
      if (t + 1 < T_STEPS)                       // warm L2 for next step
        __builtin_prefetch(gsrc + B_SZ * H_SZ, 0, 1);
      asm volatile("s_wait_asynccnt 0" ::: "memory");
    }
    __syncthreads();

    // ---- gate = sigmoid(<mem[r]+keys[m], sent[b]>)  (keys term == keys@sent^T) ----
    {
      float acc = 0.f;
      const int h0 = halfDot * 128;
      const unsigned short* mp = memB + rDot * H_SZ + h0;
      const float* kp = keys + mDot * H_SZ + h0;
      const float* sp = sentF + blDot * H_SZ + h0;
      for (int i = 0; i < 128; i += 4) {
        const unsigned long long mv = *(const unsigned long long*)(mp + i);
        const float4 kv = *(const float4*)(kp + i);
        const float4 sv = *(const float4*)(sp + i);
        acc = fmaf(bf2f((unsigned short)(mv)) + kv.x, sv.x, acc);
        acc = fmaf(bf2f((unsigned short)(mv >> 16)) + kv.y, sv.y, acc);
        acc = fmaf(bf2f((unsigned short)(mv >> 32)) + kv.z, sv.z, acc);
        acc = fmaf(bf2f((unsigned short)(mv >> 48)) + kv.w, sv.w, acc);
      }
      partF[tid] = acc;
    }
    __syncthreads();
    if (tid < 128) {
      const int m = tid >> 1, bl = tid & 1;
      const float x = partF[2 * tid] + partF[2 * tid + 1];
      const float g = 1.0f / (1.0f + __expf(-x));
      gatesOut[t * (M_SZ * B_SZ) + m * B_SZ + (bg0 + bl)] = g;
      gateF[tid] = g * mask[t * B_SZ + bg0 + bl];
      ssF[tid]   = 0.0f;
    }
    __syncthreads();

    // ---- GEMM: pre[128,256] = mem_bf16 @ U_bf16 (f32 accum) ----
    v8f acc[16];
#pragma unroll
    for (int i = 0; i < 16; ++i) acc[i] = zero8;
    const int rowA = wave * 16 + laneLo;
    for (int kt = 0; kt < 8; ++kt) {
      Frag a;   // A operand straight out of the bf16 state tile
      const unsigned short* ap = memB + rowA * H_SZ + kt * 32 + laneHi * 16;
      a.q[0] = *(const u32x4*)(ap);        // K kbase..+7
      a.q[1] = *(const u32x4*)(ap + 8);    // K kbase+8..+15
      const int kb16 = kt * 32 + laneHi * 16;
#pragma unroll
      for (int nt = 0; nt < 16; ++nt) {
        Frag b;
        const unsigned short* bp = ut + (nt * 16 + laneLo) * H_SZ + kb16;
        b.q[0] = *(const u32x4*)(bp);
        b.q[1] = *(const u32x4*)(bp + 8);
        acc[nt] = __builtin_amdgcn_wmma_f32_16x16x32_bf16(false, a.bf, false, b.bf,
                                                          (short)0, acc[nt], false, false);
      }
    }

    // ---- gated PReLU update in C-register layout; row sum-of-squares ----
#pragma unroll
    for (int j = 0; j < 8; ++j) {
      const int r = wave * 16 + laneHi * 8 + j;
      const int m = r >> 1, bl = r & 1;
      const float g = gateF[r];
      float ssl = 0.f;
#pragma unroll
      for (int nt = 0; nt < 16; ++nt) {
        const int col = nt * 16 + laneLo;
        const float pre  = acc[nt][j] + KC[m * H_SZ + col] + swF[bl * H_SZ + col];
        const float al   = prelu_a[col];
        const float cand = pre >= 0.f ? pre : al * pre;
        const float mold = bf2f(memB[r * H_SZ + col]);
        const float mnew = fmaf(g, cand, mold);
        acc[nt][j] = mnew;                 // stash un-normalized new value
        ssl = fmaf(mnew, mnew, ssl);
      }
      atomicAdd(&ssF[r], ssl);             // ds_add_f32 cross-lane row reduction
    }
    __syncthreads();

    // ---- normalize rows, write back bf16 state ----
#pragma unroll
    for (int j = 0; j < 8; ++j) {
      const int r = wave * 16 + laneHi * 8 + j;
      const float inv = 1.0f / fmaxf(sqrtf(ssF[r]), EPSV);
#pragma unroll
      for (int nt = 0; nt < 16; ++nt) {
        const int col = nt * 16 + laneLo;
        memB[r * H_SZ + col] = f2bf(acc[nt][j] * inv);
      }
    }
    __syncthreads();
  }

  // ---- epilogue: emit final memory as f32 ----
  for (int idx = tid; idx < 128 * H_SZ; idx += 256) {
    const int r = idx >> 8, h = idx & 255;
    const int m = r >> 1, b = bg0 + (r & 1);
    memOut[(m * B_SZ + b) * H_SZ + h] = bf2f(memB[idx]);
  }
}

// ---------------------------------------------------------------------------
extern "C" void kernel_launch(void* const* d_in, const int* in_sizes, int n_in,
                              void* d_out, int out_size, void* d_ws, size_t ws_size,
                              hipStream_t stream) {
  (void)in_sizes; (void)n_in; (void)out_size; (void)ws_size;
  const float* stories = (const float*)d_in[0];
  const float* mask    = (const float*)d_in[1];
  const float* keys    = (const float*)d_in[2];
  const float* U       = (const float*)d_in[3];
  const float* W       = (const float*)d_in[4];
  const float* V       = (const float*)d_in[5];
  const float* prelu_a = (const float*)d_in[6];

  float* memOut   = (float*)d_out;
  float* gatesOut = (float*)d_out + (size_t)M_SZ * B_SZ * H_SZ;

  // workspace: SW[T*B*H] f32 (64 MB) then KC[M*H] f32
  float* SW = (float*)d_ws;
  float* KC = SW + (size_t)T_STEPS * B_SZ * H_SZ;

  (void)hipFuncSetAttribute(reinterpret_cast<const void*>(sw_gemm_kernel),
                            hipFuncAttributeMaxDynamicSharedMemorySize, 131072);
  (void)hipFuncSetAttribute(reinterpret_cast<const void*>(scan_kernel),
                            hipFuncAttributeMaxDynamicSharedMemorySize, 202752);

  kc_kernel<<<M_SZ, H_SZ, 0, stream>>>(keys, V, KC);
  sw_gemm_kernel<<<(T_STEPS * B_SZ) / 128, 256, 131072, stream>>>(stories, W, SW);
  scan_kernel<<<B_SZ / 2, 256, 202752, stream>>>(stories, mask, keys, U, prelu_a,
                                                 SW, KC, memOut, gatesOut);
}
